// ReFNN2d_48241072669331
// MI455X (gfx1250) — matrix operand order repaired and checked by reading, not compiled, not run on hardware
//
#include <hip/hip_runtime.h>
#include <hip/hip_bf16.h>
#include <math.h>
#include <stdint.h>

// ---------- types ----------
typedef __attribute__((ext_vector_type(16))) __bf16 bf16x16;
typedef __attribute__((ext_vector_type(8)))  __bf16 bf16x8;
typedef __attribute__((ext_vector_type(8)))  float  f32x8;

#define HW 65536      // 256*256 points per (b,c) plane
#define LDA 136       // padded bf16 LDS row stride (mult of 8 -> 16B-aligned rows)

// =====================================================================
// conv1x1 as WMMA GEMM:  out[b,cout,p] = act( A[b,cin,p] @ W[cout,cin] + bias )
// Tile: 128 points x 128 couts per block (blockIdx.y selects cout slab).
// A tile: manual stage (transpose + fp32->bf16 convert) into LDS [pt][cin].
// W tile: bf16 pre-converted in global, layout already [cout][cin] ->
//         staged via async global->LDS DMA (ASYNCcnt), double-buffered so
//         the next K-chunk's DMA overlaps the current chunk's WMMAs.
// =====================================================================
template<typename TA, int CIN, bool SILU, bool RESID, bool OUTBF>
__global__ __launch_bounds__(256, 1)
void conv1x1_wmma(const TA* __restrict__ A, const __bf16* __restrict__ Wb,
                  const float* __restrict__ bias,
                  float* __restrict__ outF, __bf16* __restrict__ outB,
                  const float* __restrict__ resid, int coutTotal)
{
    __shared__ __bf16 lA[128 * LDA];        // [point][cin]        34 KB
    __shared__ __bf16 lW[2][128 * LDA];     // [cout][cin] x2      68 KB

    const int tid   = threadIdx.x;
    const int wave  = tid >> 5;
    const int lane  = tid & 31;
    const int g0    = blockIdx.x * 128;       // global point tile base
    const int b     = g0 >> 16;               // batch (HW points / image)
    const int p0    = g0 & (HW - 1);
    const int mbase = blockIdx.y * 128;       // cout slab base

    const __bf16* Wbase = Wb + (size_t)mbase * CIN;

    // issue async DMA of one 128x128 bf16 weight tile into lW[buf]
    auto issue_w_tile = [&](int kc, int buf) {
        unsigned long long gb =
            (unsigned long long)(uintptr_t)(Wbase + (size_t)kc * 128);
        #pragma unroll
        for (int i = 0; i < 8; ++i) {
            int seg = i * 256 + tid;              // 2048 x 16B segments
            int mm  = seg >> 4;
            int ss  = seg & 15;
            unsigned int loff =
                (unsigned int)(uintptr_t)&lW[buf][mm * LDA + ss * 8];
            unsigned int goff = (unsigned int)((mm * CIN + ss * 8) * 2);
            asm volatile("global_load_async_to_lds_b128 %0, %1, %2"
                         :: "v"(loff), "v"(goff), "s"(gb) : "memory");
        }
    };

    f32x8 acc[8];
    #pragma unroll
    for (int mt = 0; mt < 8; ++mt)
        #pragma unroll
        for (int e = 0; e < 8; ++e) acc[mt][e] = 0.0f;

    const int m_a   = wave * 16 + (lane & 15);   // A row (point within tile)
    const int khalf = (lane >> 4) * 8;           // A K sub-base per half-wave
    const int bk    = (lane >> 4) * 8;           // B K base per half-wave
    const int n     = lane & 15;                 // C/D column

    const int NC = CIN / 128;
    issue_w_tile(0, 0);

    for (int kc = 0; kc < NC; ++kc) {
        const int kbase = kc * 128;
        __syncthreads();   // everyone done computing previous chunk
        // ---- stage A tile (128 pts x 128 cin), transpose -> [pt][cin], cvt bf16
        #pragma unroll 4
        for (int i = 0; i < 64; ++i) {
            int idx = i * 256 + tid;
            int ch  = idx >> 7;
            int pp  = idx & 127;
            float v = (float)A[((size_t)(b * CIN + kbase + ch)) * HW + p0 + pp];
            lA[pp * LDA + ch] = (__bf16)v;
        }
        // prefetch next K-chunk of activations into cache (global_prefetch_b8)
        if (kc + 1 < NC) {
            int ch = tid >> 1;
            int pp = (tid & 1) * 64;
            __builtin_prefetch(&A[((size_t)(b * CIN + kbase + 128 + ch)) * HW + p0 + pp], 0, 1);
        }
        // weight DMA for this chunk must have landed
        asm volatile("s_wait_asynccnt 0" ::: "memory");
        __syncthreads();   // tile visible to all waves
        // overlap: kick off DMA for next chunk's weights into the other buffer
        if (kc + 1 < NC) issue_w_tile(kc + 1, (kc + 1) & 1);

        const __bf16* lWc = lW[kc & 1];
        #pragma unroll
        for (int ks = 0; ks < 4; ++ks) {           // 4 x K=32 per 128-chunk
            const int k0 = ks * 32;
            bf16x16 af;
            {
                const __bf16* ap0 = &lA[m_a * LDA + k0 + khalf];
                const __bf16* ap1 = ap0 + 16;
                #pragma unroll
                for (int e = 0; e < 8; ++e) { af[e] = ap0[e]; af[8 + e] = ap1[e]; }
            }
            #pragma unroll
            for (int mt = 0; mt < 8; ++mt) {
                bf16x16 bf;
                const __bf16* bp = &lWc[(mt * 16 + n) * LDA + k0 + bk];
                #pragma unroll
                for (int e = 0; e < 8; ++e) { bf[e] = bp[e]; bf[8 + e] = bp[16 + e]; }
                acc[mt] = __builtin_amdgcn_wmma_f32_16x16x32_bf16(
                    false, af, false, bf, (short)0, acc[mt], false, false);
            }
        }
    }

    // ---- epilogue: bias (+SiLU) (+residual), vectorized stores
    const int rbase = (lane >> 4) * 8;
    const int pt    = p0 + wave * 16 + rbase;      // 8 consecutive points per lane
    #pragma unroll
    for (int mt = 0; mt < 8; ++mt) {
        const int   cout = mbase + mt * 16 + n;
        const float bv   = bias[cout];
        const size_t o   = ((size_t)b * coutTotal + cout) * HW + pt;
        float vals[8];
        #pragma unroll
        for (int r = 0; r < 8; ++r) {
            float xv = acc[mt][r] + bv;
            if (SILU) xv = xv / (1.0f + __expf(-xv));
            vals[r] = xv;
        }
        if (RESID) {
            #pragma unroll
            for (int r = 0; r < 8; ++r) vals[r] += resid[o + r];
        }
        if (OUTBF) {
            bf16x8 pk;
            #pragma unroll
            for (int r = 0; r < 8; ++r) pk[r] = (__bf16)vals[r];
            *(bf16x8*)(outB + o) = pk;
        } else {
            *(float4*)(outF + o)     = make_float4(vals[0], vals[1], vals[2], vals[3]);
            *(float4*)(outF + o + 4) = make_float4(vals[4], vals[5], vals[6], vals[7]);
        }
    }
}

// =====================================================================
// 256-pt Stockham radix-2 FFT, one wave per row, LDS ping-pong.
// dir = -1: forward, +1: inverse (unnormalized). Result ends in X.
// =====================================================================
__device__ inline void fft256(float2* X, float2* Y, int lane, float dir)
{
    float2* src = X; float2* dst = Y;
    int m = 1;
    for (int t = 0; t < 8; ++t) {
        int l = 128 >> t;
        __syncthreads();
        for (int i = 0; i < 4; ++i) {
            int j = lane + 32 * i;
            int p = j >> t;
            int q = j & (m - 1);
            float ang = dir * 6.283185307179586f * (float)p / (float)(2 * l);
            float s, c;
            __sincosf(ang, &s, &c);
            float2 a  = src[q + m * p];
            float2 bb = src[q + m * (p + l)];
            float2 dif = make_float2(a.x - bb.x, a.y - bb.y);
            dst[q + 2 * m * p]     = make_float2(a.x + bb.x, a.y + bb.y);
            dst[q + 2 * m * p + m] = make_float2(c * dif.x - s * dif.y,
                                                 c * dif.y + s * dif.x);
        }
        float2* tmp = src; src = dst; dst = tmp;
        m <<= 1;
    }
    __syncthreads();
}

// forward row rfft: real 256 -> 129 complex, written transposed S1[img][wh][h]
__global__ __launch_bounds__(256) void rfft_rows_kernel(const float* __restrict__ xin,
                                                        float2* __restrict__ S1)
{
    __shared__ float2 Xs[8][256];
    __shared__ float2 Ys[8][256];
    int sub = threadIdx.x >> 5, lane = threadIdx.x & 31;
    int row = blockIdx.x * 8 + sub;
    int img = row >> 8;
    int hh  = row & 255;
    const float* src = xin + (size_t)img * HW + hh * 256;
    for (int i = 0; i < 8; ++i) {
        int col = lane + 32 * i;
        Xs[sub][col] = make_float2(src[col], 0.0f);
    }
    fft256(Xs[sub], Ys[sub], lane, -1.0f);
    for (int i = 0; i < 5; ++i) {
        int wh = lane + 32 * i;
        if (wh < 129)
            S1[((size_t)img * 129 + wh) * 256 + hh] = Xs[sub][wh];
    }
}

// column FFT + radially-masked complex filter + inverse column FFT
__global__ __launch_bounds__(256) void spectral_cols_kernel(const float2* __restrict__ S1,
        float2* __restrict__ S0, const float* __restrict__ mags,
        const float* __restrict__ phases)
{
    __shared__ float2 Xs[8][256];
    __shared__ float2 Ys[8][256];
    int sub = threadIdx.x >> 5, lane = threadIdx.x & 31;
    int col = blockIdx.x * 8 + sub;          // [0, 512*129)
    int img = col / 129;
    int wh  = col - img * 129;
    int c   = img & 127;                     // channel within image stack
    const float2* src = S1 + (size_t)col * 256;
    for (int i = 0; i < 8; ++i) Xs[sub][lane + 32 * i] = src[lane + 32 * i];
    fft256(Xs[sub], Ys[sub], lane, -1.0f);
    for (int i = 0; i < 8; ++i) {
        int r  = lane + 32 * i;
        int ky = (r < 128) ? r : r - 256;
        float2 v   = Xs[sub][r];
        float2 out = make_float2(0.0f, 0.0f);
        if (wh * wh + ky * ky <= 16384) {    // waves <= 128
            size_t fi = ((size_t)c * 256 + r) * 129 + wh;
            float g = 1.0f / (1.0f + __expf(-mags[fi]));
            float fs, fc;
            __sincosf(phases[fi], &fs, &fc);
            out = make_float2(g * (v.x * fc - v.y * fs),
                              g * (v.x * fs + v.y * fc));
        }
        Xs[sub][r] = out;
    }
    fft256(Xs[sub], Ys[sub], lane, +1.0f);   // leading __syncthreads protects writes
    for (int i = 0; i < 8; ++i) {
        int r = lane + 32 * i;
        S0[((size_t)img * 256 + r) * 129 + wh] = Xs[sub][r];
    }
}

// inverse row rfft: 129 complex (Hermitian) -> real 256, scale 1/(H*W)
__global__ __launch_bounds__(256) void irfft_rows_kernel(const float2* __restrict__ S0,
                                                         float* __restrict__ xout)
{
    __shared__ float2 Xs[8][256];
    __shared__ float2 Ys[8][256];
    int sub = threadIdx.x >> 5, lane = threadIdx.x & 31;
    int row = blockIdx.x * 8 + sub;
    int img = row >> 8;
    int hh  = row & 255;
    const float2* src = S0 + ((size_t)img * 256 + hh) * 129;
    for (int i = 0; i < 8; ++i) {
        int k = lane + 32 * i;
        float2 v;
        if (k <= 128) v = src[k];
        else { v = src[256 - k]; v.y = -v.y; }
        Xs[sub][k] = v;
    }
    fft256(Xs[sub], Ys[sub], lane, +1.0f);
    float* dst = xout + (size_t)img * HW + hh * 256;
    const float inv = 1.0f / 65536.0f;       // ortho fwd+inv overall scale
    for (int i = 0; i < 8; ++i) {
        int cidx = lane + 32 * i;
        dst[cidx] = Xs[sub][cidx].x * inv;
    }
}

// h = enc(x)*scale + pos_emb
__global__ __launch_bounds__(256) void encode_kernel(const float* __restrict__ x,
    const float* __restrict__ ew, const float* __restrict__ eb,
    const float* __restrict__ pos, float* __restrict__ h)
{
    size_t base = ((size_t)blockIdx.x * 256 + threadIdx.x) * 8;
    int b = (int)(base >> 23);
    int c = (int)((base >> 16) & 127);
    int p = (int)(base & 65535);
    float w0 = ew[c * 2 + 0], w1 = ew[c * 2 + 1], bb = eb[c];
    const float* x0 = x + ((size_t)b * 2 + 0) * HW;
    const float* x1 = x + ((size_t)b * 2 + 1) * HW;
    const float* pp = pos + (size_t)c * HW;
    float* hp = h + base;
    #pragma unroll
    for (int j = 0; j < 8; ++j)
        hp[j] = (x0[p + j] * w0 + x1[p + j] * w1 + bb) * 8.0f + pp[p + j];
}

// out = dec(h/scale)
__global__ __launch_bounds__(256) void decode_kernel(const float* __restrict__ h,
    const float* __restrict__ dw, const float* __restrict__ db, float* __restrict__ out)
{
    int idx = blockIdx.x * 256 + threadIdx.x;   // (b,p)
    int b = idx >> 16;
    int p = idx & 65535;
    const float* hp = h + (size_t)b * 128 * HW + p;
    float s0 = 0.0f, s1 = 0.0f;
    for (int c = 0; c < 128; ++c) {
        float v = hp[(size_t)c * HW];
        s0 += v * dw[c];
        s1 += v * dw[128 + c];
    }
    out[((size_t)b * 2 + 0) * HW + p] = s0 * 0.125f + db[0];
    out[((size_t)b * 2 + 1) * HW + p] = s1 * 0.125f + db[1];
}

// fp32 -> bf16 weight conversion (once per launch; deterministic)
__global__ __launch_bounds__(256) void cvt_bf16_kernel(const float* __restrict__ s,
                                                       __bf16* __restrict__ d, int n)
{
    int i = blockIdx.x * 256 + threadIdx.x;
    if (i < n) d[i] = (__bf16)s[i];
}

// =====================================================================
extern "C" void kernel_launch(void* const* d_in, const int* in_sizes, int n_in,
                              void* d_out, int out_size, void* d_ws, size_t ws_size,
                              hipStream_t stream)
{
    (void)in_sizes; (void)n_in; (void)out_size; (void)ws_size;
    const float* x       = (const float*)d_in[0];
    const float* enc_w   = (const float*)d_in[1];
    const float* enc_b   = (const float*)d_in[2];
    const float* pos_emb = (const float*)d_in[3];
    const float* mlp_w1  = (const float*)d_in[4];
    const float* mlp_b1  = (const float*)d_in[5];
    const float* mlp_w2  = (const float*)d_in[6];
    const float* mlp_b2  = (const float*)d_in[7];
    const float* mags    = (const float*)d_in[8];
    const float* phases  = (const float*)d_in[9];
    const float* oxo_w   = (const float*)d_in[10];
    const float* oxo_b   = (const float*)d_in[11];
    const float* dec_w   = (const float*)d_in[12];
    const float* dec_b   = (const float*)d_in[13];
    float* out = (float*)d_out;

    // workspace layout (total 808,583,168 B)
    char*   ws  = (char*)d_ws;
    float*  h   = (float*)(ws);                                  // 128 MiB
    __bf16* x1  = (__bf16*)(ws + (size_t)134217728);             // 256 MiB bf16
    float*  x2  = (float*)(ws + (size_t)402653184);              // 128 MiB
    float2* S1  = (float2*)(ws + (size_t)536870912);             // 129 MiB (img,wh,h)
    float2* S0  = (float2*)(ws + (size_t)672137216);             // 129 MiB (img,h,wh)
    __bf16* w1b = (__bf16*)(ws + (size_t)807403520);             // 512 KiB
    __bf16* w2b = (__bf16*)(ws + (size_t)807927808);             // 512 KiB
    __bf16* oxb = (__bf16*)(ws + (size_t)808452096);             // 128 KiB

    // pre-convert weights to bf16 (enables async W-tile DMA in the GEMMs)
    cvt_bf16_kernel<<<1024, 256, 0, stream>>>(mlp_w1, w1b, 4 * 512 * 128);
    cvt_bf16_kernel<<<1024, 256, 0, stream>>>(mlp_w2, w2b, 4 * 128 * 512);
    cvt_bf16_kernel<<<256,  256, 0, stream>>>(oxo_w,  oxb, 4 * 128 * 128);

    encode_kernel<<<16384, 256, 0, stream>>>(x, enc_w, enc_b, pos_emb, h);

    for (int L = 0; L < 4; ++L) {
        // x1 = silu(h @ w1^T + b1)  (128 -> 512), bf16 out
        conv1x1_wmma<float, 128, true, false, true><<<dim3(2048, 4), 256, 0, stream>>>(
            h, w1b + (size_t)L * 512 * 128, mlp_b1 + L * 512,
            nullptr, x1, nullptr, 512);
        // x2 = x1 @ w2^T + b2  (512 -> 128), fp32 out for FFT
        conv1x1_wmma<__bf16, 512, false, false, false><<<dim3(2048, 1), 256, 0, stream>>>(
            x1, w2b + (size_t)L * 128 * 512, mlp_b2 + L * 128,
            x2, nullptr, nullptr, 128);
        // spectral conv: rfft2 -> masked complex filter -> irfft2 (in x2)
        rfft_rows_kernel<<<16384, 256, 0, stream>>>(x2, S1);
        spectral_cols_kernel<<<8256, 256, 0, stream>>>(S1, S0,
            mags   + (size_t)L * 128 * 256 * 129,
            phases + (size_t)L * 128 * 256 * 129);
        irfft_rows_kernel<<<16384, 256, 0, stream>>>(S0, x2);
        // h = x2 @ oxo^T + b + h  (128 -> 128, residual, in-place)
        conv1x1_wmma<float, 128, false, true, false><<<dim3(2048, 1), 256, 0, stream>>>(
            x2, oxb + (size_t)L * 128 * 128, oxo_b + L * 128,
            h, nullptr, h, 128);
    }

    decode_kernel<<<1024, 256, 0, stream>>>(h, dec_w, dec_b, out);
}
